// MSAColumnGlobalAttention_64707977281628
// MI455X (gfx1250) — matrix operand order repaired and checked by reading, not compiled, not run on hardware
//
#include <hip/hip_runtime.h>
#include <hip/hip_bf16.h>

// Problem constants from the reference
#define S_LEN 2048
#define R_LEN 512
#define C_IN  64
#define NH    8
#define CHD   16
#define HC    128       // NH*CHD
#define INF_F 1.0e9f
#define EPS_F 1.0e-10f
#define LNEPS 1.0e-5f
#define NCHUNK (S_LEN / 64)   // 32 chunks of 64 rows

typedef __bf16 bf16;
typedef __attribute__((ext_vector_type(16))) __bf16 v16bf;
typedef __attribute__((ext_vector_type(8)))  float  v8f;
typedef int v4i_ __attribute__((vector_size(16)));   // matches async builtin param type

#define AS1_ __attribute__((address_space(1)))
#define AS3_ __attribute__((address_space(3)))

union Frag {
    uint4 u[2];
    v16bf v;
};
union U4H {
    uint4 u;
    bf16  h[8];
};
union PK4 {
    bf16  h[4];
    uint2 u;
};

// ---- gfx1250 async global->LDS copy (ASYNCcnt-tracked), with fallbacks ----
// Param types per round-2 diagnostic: (int4 __device__*, int4 __shared__*, imm, imm).
// Integer round-trip casts: flat==AS1 for global; flat LDS truncates to the
// 32-bit LDS offset (ISA 10.2) -> semantically correct on gfx1250.
#if __has_builtin(__builtin_amdgcn_global_load_async_to_lds_b128)
  #define HAVE_ASYNC 1
  #define ASYNC_CP16(gp, lp)                                                   \
      __builtin_amdgcn_global_load_async_to_lds_b128(                          \
          (AS1_ v4i_*)(unsigned long long)(gp),                                \
          (AS3_ v4i_*)(unsigned)(unsigned long long)(lp), 0, 0)
#elif __has_builtin(__builtin_amdgcn_load_to_lds)
  #define HAVE_ASYNC 1
  #define ASYNC_CP16(gp, lp)                                                   \
      __builtin_amdgcn_load_to_lds(                                            \
          (AS1_ v4i_*)(unsigned long long)(gp),                                \
          (AS3_ v4i_*)(unsigned)(unsigned long long)(lp), 16, 0, 0)
#else
  #define HAVE_ASYNC 0
  #define ASYNC_CP16(gp, lp) (*(float4*)(lp) = *(const float4*)(gp))
#endif

#if HAVE_ASYNC
  #if __has_builtin(__builtin_amdgcn_s_wait_asynccnt)
    #define WAIT_ASYNC(n) __builtin_amdgcn_s_wait_asynccnt(n)
  #else
    #define WAIT_ASYNC(n) asm volatile("s_wait_asynccnt %0" ::"n"(n) : "memory")
  #endif
#else
  #define WAIT_ASYNC(n)
#endif

__global__ __launch_bounds__(256) void msa_col_global_attn_kernel(
    const float* __restrict__ mp,   const float* __restrict__ maskp,
    const float* __restrict__ lnw,  const float* __restrict__ lnb,
    const float* __restrict__ Wq,   const float* __restrict__ Wk,
    const float* __restrict__ Wv,   const float* __restrict__ Wg,
    const float* __restrict__ bg,   const float* __restrict__ Wo,
    const float* __restrict__ bo,   float* __restrict__ outp)
{
    // ---- LDS ----
    __shared__ float2 sStats[S_LEN];                                  // 16 KB (mu, rstd)
    __shared__ __align__(16) float sMraw[2][64][C_IN];                // 32 KB raw f32 staging
    __shared__ __align__(16) bf16 sMn[128][C_IN];                     // 16 KB LN'd tile
    __shared__ __align__(16) bf16 sK[64][CHD];                        // 2 KB
    __shared__ __align__(16) bf16 sV[64][CHD];                        // 2 KB
    __shared__ float sP[NH][64];                                      // 2 KB softmax probs
    __shared__ __align__(16) bf16 sWkT[CHD][C_IN];                    // 2 KB  WkT[n][c]
    __shared__ __align__(16) bf16 sWvT[CHD][C_IN];                    // 2 KB
    __shared__ __align__(16) bf16 sWgT[HC][C_IN];                     // 16 KB WgT[n][c]
    __shared__ __align__(16) bf16 sWoT[C_IN][HC];                     // 16 KB WoT[c][n]
    __shared__ __align__(16) bf16 sOg[8][16][32];                     // 8 KB per-wave og chunk
    __shared__ float sLnW[C_IN], sLnB[C_IN], sBo[C_IN], sBg[HC];
    __shared__ float sQ[HC];                                          // q @ Wq (scaled)
    __shared__ float sO[NH][CHD];                                     // pooled attention out
    __shared__ float sQred[16][C_IN];                                 // 4 KB pooling partials
    __shared__ float sCnt[256];
    __shared__ float sCountS;

    const int r    = blockIdx.x;
    const int tid  = threadIdx.x;
    const int team = tid >> 4;          // 16 row-teams of 16 lanes
    const int l16  = tid & 15;
    const int wave = tid >> 5;          // 8 waves (wave32)
    const int lane = tid & 31;

    // Issue one 64-row chunk copy (16 KB): 1024 16-byte segments, 4 per thread.
    auto issue_chunk = [&](int ch, int par) {
        #pragma unroll
        for (int it = 0; it < 4; ++it) {
            int seg = tid + it * 256;
            int row = seg >> 4;
            int c4  = (seg & 15) << 2;
            const float* gp = mp + ((size_t)(ch * 64 + row) * R_LEN + r) * C_IN + c4;
            float* lp = &sMraw[par][row][c4];
            ASYNC_CP16(gp, lp);
        }
    };

    // LN-convert one 16-row step of a staged chunk into sMn (packed b64 store)
    auto ln_convert = [&](int p, int sl, int s, int dr) {
        const float4 x = *reinterpret_cast<const float4*>(&sMraw[p][sl][l16 * 4]);
        float2 st = sStats[s];
        int c = l16 * 4;
        PK4 pk;
        pk.h[0] = (bf16)((x.x - st.x) * st.y * sLnW[c + 0] + sLnB[c + 0]);
        pk.h[1] = (bf16)((x.y - st.x) * st.y * sLnW[c + 1] + sLnB[c + 1]);
        pk.h[2] = (bf16)((x.z - st.x) * st.y * sLnW[c + 2] + sLnB[c + 2]);
        pk.h[3] = (bf16)((x.w - st.x) * st.y * sLnW[c + 3] + sLnB[c + 3]);
        *reinterpret_cast<uint2*>(&sMn[dr][c]) = pk.u;
    };

    // ---- Stage 0: weights -> LDS (bf16, transposed for B-fragment loads) ----
    for (int idx = tid; idx < CHD * C_IN; idx += 256) {
        int n = idx / C_IN, c = idx % C_IN;
        sWkT[n][c] = (bf16)Wk[c * CHD + n];
        sWvT[n][c] = (bf16)Wv[c * CHD + n];
    }
    for (int idx = tid; idx < HC * C_IN; idx += 256) {
        int n = idx / C_IN, c = idx % C_IN;
        sWgT[n][c] = (bf16)Wg[c * HC + n];
    }
    for (int idx = tid; idx < C_IN * HC; idx += 256) {
        int c = idx / HC, n = idx % HC;
        sWoT[c][n] = (bf16)Wo[n * C_IN + c];
    }
    if (tid < C_IN) { sLnW[tid] = lnw[tid]; sLnB[tid] = lnb[tid]; sBo[tid] = bo[tid]; }
    if (tid < HC)   { sBg[tid] = bg[tid]; }
    __syncthreads();

    // ================= Pass 1: LN stats + masked mean pooling =================
    float qloc0 = 0.f, qloc1 = 0.f, qloc2 = 0.f, qloc3 = 0.f;
    float cntloc = 0.f;
    issue_chunk(0, 0);
    for (int ch = 0; ch < NCHUNK; ++ch) {
        const int p = ch & 1;
        if (ch + 1 < NCHUNK) { issue_chunk(ch + 1, p ^ 1); WAIT_ASYNC(4); }
        else                 { WAIT_ASYNC(0); }
        __syncthreads();
        #pragma unroll
        for (int step = 0; step < 4; ++step) {
            int sl = step * 16 + team;
            int s  = ch * 64 + sl;
            const float4 x = *reinterpret_cast<const float4*>(&sMraw[p][sl][l16 * 4]);
            float sum = x.x + x.y + x.z + x.w;
            float sq  = x.x * x.x + x.y * x.y + x.z * x.z + x.w * x.w;
            #pragma unroll
            for (int d = 8; d >= 1; d >>= 1) {
                sum += __shfl_xor(sum, d, 16);
                sq  += __shfl_xor(sq, d, 16);
            }
            float mu   = sum * (1.f / 64.f);
            float var  = sq * (1.f / 64.f) - mu * mu;
            float rstd = rsqrtf(var + LNEPS);
            if (l16 == 0) sStats[s] = make_float2(mu, rstd);
            float mk = maskp[(size_t)s * R_LEN + r];
            cntloc += mk;
            int c = l16 * 4;
            qloc0 += ((x.x - mu) * rstd * sLnW[c + 0] + sLnB[c + 0]) * mk;
            qloc1 += ((x.y - mu) * rstd * sLnW[c + 1] + sLnB[c + 1]) * mk;
            qloc2 += ((x.z - mu) * rstd * sLnW[c + 2] + sLnB[c + 2]) * mk;
            qloc3 += ((x.w - mu) * rstd * sLnW[c + 3] + sLnB[c + 3]) * mk;
        }
        __syncthreads();   // chunk consumed; staging buffer may be re-issued
    }
    sQred[team][l16 * 4 + 0] = qloc0;
    sQred[team][l16 * 4 + 1] = qloc1;
    sQred[team][l16 * 4 + 2] = qloc2;
    sQred[team][l16 * 4 + 3] = qloc3;
    sCnt[tid] = cntloc;
    __syncthreads();
    if (tid == 0) {
        float c = 0.f;
        for (int i = 0; i < 256; ++i) c += sCnt[i];
        sCountS = c * (1.f / 16.f);   // each team lane duplicated the count
    }
    if (tid < C_IN) {
        float a = 0.f;
        #pragma unroll
        for (int t = 0; t < 16; ++t) a += sQred[t][tid];
        sQred[0][tid] = a;            // qsum[c]
    }
    __syncthreads();
    if (tid < HC) {                   // q = (qsum/count) @ Wq * CH^-0.5
        float inv = 0.25f / (sCountS + EPS_F);
        float a = 0.f;
        #pragma unroll 8
        for (int c = 0; c < C_IN; ++c) a += sQred[0][c] * Wq[c * HC + tid];
        sQ[tid] = a * inv;
    }
    __syncthreads();

    // ====== Pass 2: k/v via WMMA + online (flash) softmax of pooled attn ======
    float runM = -3.0e38f, runL = 0.f, oacc = 0.f;
    issue_chunk(0, 0);
    for (int ch = 0; ch < NCHUNK; ++ch) {
        const int p  = ch & 1;
        const int tb = ch * 64;
        if (ch + 1 < NCHUNK) { issue_chunk(ch + 1, p ^ 1); WAIT_ASYNC(4); }
        else                 { WAIT_ASYNC(0); }
        __syncthreads();
        #pragma unroll
        for (int step = 0; step < 4; ++step) {
            int sl = step * 16 + team;
            ln_convert(p, sl, tb + sl, sl);
        }
        __syncthreads();

        // waves 0-3: k tiles; waves 4-7: v tiles (16 rows each)
        {
            const int isV = wave >> 2;
            const int rw  = (wave & 3) * 16;
            v8f acc = {0.f, 0.f, 0.f, 0.f, 0.f, 0.f, 0.f, 0.f};
            #pragma unroll
            for (int ks = 0; ks < 2; ++ks) {
                const int koff = ks * 32;
                Frag A, B;
                int row = rw + (lane & 15);
                int c0  = koff + ((lane >> 4) << 3);
                A.u[0] = *reinterpret_cast<const uint4*>(&sMn[row][c0]);
                A.u[1] = *reinterpret_cast<const uint4*>(&sMn[row][c0 + 16]);
                const bf16(*WT)[C_IN] = isV ? sWvT : sWkT;
                int n  = lane & 15;
                int cb = koff + ((lane >> 4) << 3);
                B.u[0] = *reinterpret_cast<const uint4*>(&WT[n][cb]);
                B.u[1] = *reinterpret_cast<const uint4*>(&WT[n][cb + 16]);
                acc = __builtin_amdgcn_wmma_f32_16x16x32_bf16(
                    false, A.v, false, B.v, (short)0, acc, false, false);
            }
            bf16(*dst)[CHD] = isV ? sV : sK;
            #pragma unroll
            for (int i = 0; i < 8; ++i) {
                int row = rw + ((lane < 16) ? i : i + 8);
                dst[row][lane & 15] = (bf16)acc[i];
            }
        }
        __syncthreads();

        // wave h owns head h: logits + online softmax update
        {
            const int h = wave;
            float lg[2];
            #pragma unroll
            for (int half = 0; half < 2; ++half) {
                int sl = half * 32 + lane;
                int s  = tb + sl;
                float mk = maskp[(size_t)s * R_LEN + r];
                float a  = INF_F * (mk - 1.f);
                U4H k0, k1;
                k0.u = *reinterpret_cast<const uint4*>(&sK[sl][0]);
                k1.u = *reinterpret_cast<const uint4*>(&sK[sl][8]);
                #pragma unroll
                for (int c = 0; c < 8; ++c) {
                    a += sQ[h * CHD + c]     * (float)k0.h[c];
                    a += sQ[h * CHD + c + 8] * (float)k1.h[c];
                }
                lg[half] = a;
            }
            float tmax = fmaxf(lg[0], lg[1]);
            #pragma unroll
            for (int d = 16; d >= 1; d >>= 1) tmax = fmaxf(tmax, __shfl_xor(tmax, d, 32));
            float newM = fmaxf(runM, tmax);
            float sc   = __expf(runM - newM);
            float p0   = __expf(lg[0] - newM);
            float p1   = __expf(lg[1] - newM);
            sP[h][lane]      = p0;
            sP[h][lane + 32] = p1;
            float ps = p0 + p1;
            #pragma unroll
            for (int d = 16; d >= 1; d >>= 1) ps += __shfl_xor(ps, d, 32);
            runL = runL * sc + ps;
            runM = newM;
            float a = 0.f;
            int ch16 = lane & 15;
            for (int sl = 0; sl < 64; ++sl) a += sP[h][sl] * (float)sV[sl][ch16];
            oacc = oacc * sc + a;      // lanes >=16 duplicate harmlessly
        }
        __syncthreads();
    }
    if (lane < 16) sO[wave][lane] = oacc / (runL + EPS_F);
    __syncthreads();

    // ========= Pass 3: gating + output projection (128-row macro-tiles) =========
    issue_chunk(0, 0);
    for (int ch = 0; ch < NCHUNK; ++ch) {
        const int p = ch & 1;
        if (ch + 1 < NCHUNK) { issue_chunk(ch + 1, p ^ 1); WAIT_ASYNC(4); }
        else                 { WAIT_ASYNC(0); }
        __syncthreads();
        #pragma unroll
        for (int step = 0; step < 4; ++step) {
            int sl = step * 16 + team;
            ln_convert(p, sl, ch * 64 + sl, (ch & 1) * 64 + sl);
        }
        __syncthreads();
        if ((ch & 1) == 0) continue;   // need both halves of the 128-row tile

        const int tb = (ch - 1) * 64;
        const int rw = wave * 16;
        v8f outacc[4];
        #pragma unroll
        for (int ct = 0; ct < 4; ++ct)
            outacc[ct] = (v8f){0.f, 0.f, 0.f, 0.f, 0.f, 0.f, 0.f, 0.f};

        #pragma unroll
        for (int jp = 0; jp < 4; ++jp) {
            // two 16-wide n-chunks of g -> og (bf16, A-layout staging in LDS)
            #pragma unroll
            for (int jj = 0; jj < 2; ++jj) {
                const int j = jp * 2 + jj;   // head index == n-tile index
                v8f g = {0.f, 0.f, 0.f, 0.f, 0.f, 0.f, 0.f, 0.f};
                #pragma unroll
                for (int ks = 0; ks < 2; ++ks) {
                    const int koff = ks * 32;
                    Frag A, B;
                    int row = rw + (lane & 15);
                    int c0  = koff + ((lane >> 4) << 3);
                    A.u[0] = *reinterpret_cast<const uint4*>(&sMn[row][c0]);
                    A.u[1] = *reinterpret_cast<const uint4*>(&sMn[row][c0 + 16]);
                    int n  = j * 16 + (lane & 15);
                    int cb = koff + ((lane >> 4) << 3);
                    B.u[0] = *reinterpret_cast<const uint4*>(&sWgT[n][cb]);
                    B.u[1] = *reinterpret_cast<const uint4*>(&sWgT[n][cb + 16]);
                    g = __builtin_amdgcn_wmma_f32_16x16x32_bf16(
                        false, A.v, false, B.v, (short)0, g, false, false);
                }
                float bgv = sBg[j * 16 + (lane & 15)];
                float ofc = sO[j][lane & 15];
                #pragma unroll
                for (int i = 0; i < 8; ++i) {
                    float gv = g[i] + bgv;
                    float sg = __frcp_rn(1.f + __expf(-gv));
                    int row  = (lane < 16) ? i : i + 8;
                    sOg[wave][row][jj * 16 + (lane & 15)] = (bf16)(ofc * sg);
                }
            }
            // out += og(16x32 K-chunk) @ Wo
            Frag A;
            {
                int row = lane & 15;
                int k0  = (lane >> 4) << 3;
                A.u[0] = *reinterpret_cast<const uint4*>(&sOg[wave][row][k0]);
                A.u[1] = *reinterpret_cast<const uint4*>(&sOg[wave][row][k0 + 16]);
            }
            #pragma unroll
            for (int ct = 0; ct < 4; ++ct) {
                Frag B;
                int cc = ct * 16 + (lane & 15);
                int nb = jp * 32 + ((lane >> 4) << 3);
                B.u[0] = *reinterpret_cast<const uint4*>(&sWoT[cc][nb]);
                B.u[1] = *reinterpret_cast<const uint4*>(&sWoT[cc][nb + 16]);
                outacc[ct] = __builtin_amdgcn_wmma_f32_16x16x32_bf16(
                    false, A.v, false, B.v, (short)0, outacc[ct], false, false);
            }
        }
        // store: out[s][r][c] (+bo)
        #pragma unroll
        for (int ct = 0; ct < 4; ++ct) {
            float bov = sBo[ct * 16 + (lane & 15)];
            #pragma unroll
            for (int i = 0; i < 8; ++i) {
                int srow = tb + rw + ((lane < 16) ? i : i + 8);
                outp[((size_t)srow * R_LEN + r) * C_IN + ct * 16 + (lane & 15)] =
                    outacc[ct][i] + bov;
            }
        }
        // end-of-tile barrier is provided by the staging barrier of the next chunk
    }
}

extern "C" void kernel_launch(void* const* d_in, const int* in_sizes, int n_in,
                              void* d_out, int out_size, void* d_ws, size_t ws_size,
                              hipStream_t stream) {
    const float* m    = (const float*)d_in[0];
    const float* mask = (const float*)d_in[1];
    const float* lnw  = (const float*)d_in[2];
    const float* lnb  = (const float*)d_in[3];
    const float* Wq   = (const float*)d_in[4];
    const float* Wk   = (const float*)d_in[5];
    const float* Wv   = (const float*)d_in[6];
    const float* Wg   = (const float*)d_in[7];
    const float* bg   = (const float*)d_in[8];
    const float* Wo   = (const float*)d_in[9];
    const float* bo   = (const float*)d_in[10];
    float* out        = (float*)d_out;

    msa_col_global_attn_kernel<<<dim3(R_LEN), dim3(256), 0, stream>>>(
        m, mask, lnw, lnb, Wq, Wk, Wv, Wg, bg, Wo, bo, out);
}
